// BasicGraphClassifier_395136991531
// MI455X (gfx1250) — matrix operand order repaired and checked by reading, not compiled, not run on hardware
//
#include <hip/hip_runtime.h>
#include <hip/hip_bf16.h>

#define N_NODES     50000
#define N_EDGES     800000
#define IN_DIM      128
#define HID         256
#define NUM_CLASSES 10
#define NUM_GRAPHS  64

typedef __attribute__((ext_vector_type(16))) _Float16 v16h;
typedef __attribute__((ext_vector_type(8)))  _Float16 v8h;
typedef __attribute__((ext_vector_type(8)))  float    v8f;

union V16H { v16h v; v8h h[2]; };

// ---------------- elementwise helpers ----------------

__global__ void zero_f32(float* __restrict__ p, int n) {
  for (int i = blockIdx.x * blockDim.x + threadIdx.x; i < n;
       i += gridDim.x * blockDim.x)
    p[i] = 0.0f;
}

__global__ void cvt_f16(const float* __restrict__ s, _Float16* __restrict__ d, int n) {
  for (int i = blockIdx.x * blockDim.x + threadIdx.x; i < n;
       i += gridDim.x * blockDim.x)
    d[i] = (_Float16)s[i];
}

// zh = f16(h + agg)   (the "z = x + sum_neighbors" of GINConv, narrowed for WMMA)
__global__ void add_cvt_f16(const float* __restrict__ a, const float* __restrict__ b,
                            _Float16* __restrict__ d, int n) {
  for (int i = blockIdx.x * blockDim.x + threadIdx.x; i < n;
       i += gridDim.x * blockDim.x)
    d[i] = (_Float16)(a[i] + b[i]);
}

// ---------------- edge scatter-add (segment_sum over edges) ----------------
// agg[dst[e], :] += h[src[e], :]   one thread = one edge x 4 floats (b128 read)
__global__ void scatter_add(const float* __restrict__ h,
                            const int* __restrict__ src,
                            const int* __restrict__ dst,
                            float* __restrict__ agg, int D4) {
  const int total = N_EDGES * D4;
  for (int i = blockIdx.x * blockDim.x + threadIdx.x; i < total;
       i += gridDim.x * blockDim.x) {
    const int e = i / D4;
    const int d = i - e * D4;
    const int s = src[e];
    const int t = dst[e];
    const float4 v = ((const float4*)(h + (size_t)s * (D4 * 4)))[d];
    float* p = agg + (size_t)t * (D4 * 4) + d * 4;
    atomicAdd(p + 0, v.x);
    atomicAdd(p + 1, v.y);
    atomicAdd(p + 2, v.z);
    atomicAdd(p + 3, v.w);
  }
}

__global__ void count_nodes(const int* __restrict__ batch, float* __restrict__ cnts) {
  for (int i = blockIdx.x * blockDim.x + threadIdx.x; i < N_NODES;
       i += gridDim.x * blockDim.x)
    atomicAdd(&cnts[batch[i]], 1.0f);
}

// ---------------- WMMA GEMM:  out[M,Kout] = A[M,Kin] * W[Kout,Kin]^T + bias ----
// Each wave owns a 16(M) x 64(N) output strip = 4 accumulators; the A fragment
// is loaded once per K-step and reused for 4 WMMAs (4x less A traffic, and A is
// the big 50000-row stream; B is the small resident weight matrix).
// MODE 0: relu, f16 output (feeds next WMMA stage)
// MODE 1: f32 output (h1, needed in f32 for next conv's z = h1 + agg)
// MODE 2: fused global mean-pool numerator: atomicAdd into sums[batch[m]*HID+n]
template <int MODE>
__global__ __launch_bounds__(256)
void gemm_wmma(const _Float16* __restrict__ A, const _Float16* __restrict__ W,
               const float* __restrict__ bias, void* __restrict__ outp,
               const int* __restrict__ batch, int M, int Kin, int Kout) {
  const int lane = threadIdx.x & 31;
  const int half = lane >> 4;            // 0: lanes 0-15, 1: lanes 16-31
  const int l15  = lane & 15;
  const int nStrips = Kout >> 6;         // 64 output cols per wave
  const int tiles   = (M >> 4) * nStrips;
  const int tile    = blockIdx.x * (blockDim.x >> 5) + (threadIdx.x >> 5);
  if (tile >= tiles) return;             // wave-uniform: EXEC stays all-ones
  const int mTile = tile / nStrips;
  const int strip = tile - mTile * nStrips;

  // A fragment rows: lane&15 (both lane halves hold the same row, lanes 16-31 hold K 8..15, 24..31)
  const _Float16* __restrict__ Ap = A + (size_t)((mTile << 4) + l15) * Kin;
  // B fragments: column (output channel) per lane; W is [Kout, Kin] row-major,
  // i.e. exactly K-major per output column as WMMA's B wants.
  const _Float16* __restrict__ Bp0 = W + (size_t)((strip << 6) + l15) * Kin;

  v8f c[4] = {{}, {}, {}, {}};
  for (int kb = 0; kb < Kin; kb += 32) {
    V16H a;
    // 16-bit A 16x32 layout: lanes 0-15 VGPR0-3 = K 0..7, VGPR4-7 = K 16..23;
    //                        lanes 16-31 VGPR0-3 = K 8..15, VGPR4-7 = K 24..31.
    a.h[0] = *(const v8h*)(Ap + kb + half * 8);
    a.h[1] = *(const v8h*)(Ap + kb + 16 + half * 8);
    const int bo = kb + half * 8;        // B: same K order as A (lanes 0-15 K 0..7,16..23; 16-31 K 8..15,24..31)
#pragma unroll
    for (int j = 0; j < 4; ++j) {
      V16H b;
      const _Float16* __restrict__ Bp = Bp0 + (size_t)(j << 4) * Kin;
      b.h[0] = *(const v8h*)(Bp + bo);
      b.h[1] = *(const v8h*)(Bp + bo + 16);
      c[j] = __builtin_amdgcn_wmma_f32_16x16x32_f16(
          /*neg_a=*/false, a.v, /*neg_b=*/false, b.v,
          /*c_mod=*/(short)0, c[j], /*reuse_a=*/false, /*reuse_b=*/false);
    }
  }

  // D layout: VGPR r -> M = r + 8*half, N = lane&15
  const int m0 = (mTile << 4) + half * 8;

  if (MODE == 2) {
    int g[8];
#pragma unroll
    for (int r = 0; r < 8; ++r) g[r] = batch[m0 + r];
    float* sums = (float*)outp;
#pragma unroll
    for (int j = 0; j < 4; ++j) {
      const int n = (strip << 6) + (j << 4) + l15;
      const float bv = bias[n];
#pragma unroll
      for (int r = 0; r < 8; ++r)
        atomicAdd(&sums[g[r] * HID + n], c[j][r] + bv);
    }
  } else {
#pragma unroll
    for (int j = 0; j < 4; ++j) {
      const int n = (strip << 6) + (j << 4) + l15;
      const float bv = bias[n];
      if (MODE == 0) {
        _Float16* out = (_Float16*)outp;
#pragma unroll
        for (int r = 0; r < 8; ++r) {
          float v = fmaxf(c[j][r] + bv, 0.0f);
          out[(size_t)(m0 + r) * Kout + n] = (_Float16)v;
        }
      } else {
        float* out = (float*)outp;
#pragma unroll
        for (int r = 0; r < 8; ++r)
          out[(size_t)(m0 + r) * Kout + n] = c[j][r] + bv;
      }
    }
  }
}

// ---------------- tiny classifier: out[g,c] = mean_repr(g) . Wc[c] + bc ------
__global__ void classify(const float* __restrict__ sums, const float* __restrict__ cnts,
                         const float* __restrict__ Wc, const float* __restrict__ bc,
                         float* __restrict__ out) {
  const int t = blockIdx.x * blockDim.x + threadIdx.x;
  if (t >= NUM_GRAPHS * NUM_CLASSES) return;
  const int g  = t / NUM_CLASSES;
  const int cl = t - g * NUM_CLASSES;
  const float inv = 1.0f / fmaxf(cnts[g], 1.0f);
  float acc = bc[cl];
  for (int d = 0; d < HID; ++d)
    acc += (sums[g * HID + d] * inv) * Wc[cl * HID + d];
  out[t] = acc;
}

// ---------------- host-side orchestration ----------------

extern "C" void kernel_launch(void* const* d_in, const int* in_sizes, int n_in,
                              void* d_out, int out_size, void* d_ws, size_t ws_size,
                              hipStream_t stream) {
  (void)in_sizes; (void)n_in; (void)out_size; (void)ws_size;

  const float* x    = (const float*)d_in[0];
  const int*   ei   = (const int*)d_in[1];     // [2, E]
  const int*   batc = (const int*)d_in[2];     // [N]
  const float* W1a  = (const float*)d_in[3];
  const float* b1a  = (const float*)d_in[4];
  const float* W1b  = (const float*)d_in[5];
  const float* b1b  = (const float*)d_in[6];
  const float* W2a  = (const float*)d_in[7];
  const float* b2a  = (const float*)d_in[8];
  const float* W2b  = (const float*)d_in[9];
  const float* b2b  = (const float*)d_in[10];
  const float* Wc   = (const float*)d_in[11];
  const float* bc   = (const float*)d_in[12];
  float* out = (float*)d_out;

  const int* src = ei;
  const int* dst = ei + N_EDGES;

  // ---- workspace carve-up (256B aligned) ----
  char* ws = (char*)d_ws;
  size_t off = 0;
  auto take = [&](size_t bytes) -> char* {
    char* p = ws + off;
    off = (off + bytes + 255) & ~(size_t)255;
    return p;
  };
  _Float16* Wh1a = (_Float16*)take((size_t)HID * IN_DIM * 2);
  _Float16* Wh1b = (_Float16*)take((size_t)HID * HID * 2);
  _Float16* Wh2a = (_Float16*)take((size_t)HID * HID * 2);
  _Float16* Wh2b = (_Float16*)take((size_t)HID * HID * 2);
  float*    sums = (float*)take((size_t)NUM_GRAPHS * HID * 4);
  float*    cnts = (float*)take((size_t)NUM_GRAPHS * 4);
  float*    agg  = (float*)take((size_t)N_NODES * HID * 4);   // reused both layers
  _Float16* zh   = (_Float16*)take((size_t)N_NODES * HID * 2);
  _Float16* th   = (_Float16*)take((size_t)N_NODES * HID * 2);
  float*    h1   = (float*)take((size_t)N_NODES * HID * 4);

  const int TB = 256;
  auto gs = [&](int n) { int b = (n + TB - 1) / TB; return b < 16384 ? b : 16384; };

  // ---- weights -> f16 (row-major [Kout,Kin] == WMMA B layout) ----
  cvt_f16<<<gs(HID * IN_DIM), TB, 0, stream>>>(W1a, Wh1a, HID * IN_DIM);
  cvt_f16<<<gs(HID * HID),    TB, 0, stream>>>(W1b, Wh1b, HID * HID);
  cvt_f16<<<gs(HID * HID),    TB, 0, stream>>>(W2a, Wh2a, HID * HID);
  cvt_f16<<<gs(HID * HID),    TB, 0, stream>>>(W2b, Wh2b, HID * HID);

  // 16x64 strip per wave, 8 waves per block
  const int gemm_blocks = ((N_NODES / 16) * (HID / 64) + 7) / 8;

  // ================= GINConv 1 (IN_DIM -> HID) =================
  zero_f32<<<gs(N_NODES * IN_DIM), TB, 0, stream>>>(agg, N_NODES * IN_DIM);
  scatter_add<<<8192, TB, 0, stream>>>(x, src, dst, agg, IN_DIM / 4);
  add_cvt_f16<<<gs(N_NODES * IN_DIM), TB, 0, stream>>>(x, agg, zh, N_NODES * IN_DIM);
  gemm_wmma<0><<<gemm_blocks, TB, 0, stream>>>(zh, Wh1a, b1a, th, nullptr,
                                               N_NODES, IN_DIM, HID);
  gemm_wmma<1><<<gemm_blocks, TB, 0, stream>>>(th, Wh1b, b1b, h1, nullptr,
                                               N_NODES, HID, HID);

  // ================= GINConv 2 (HID -> HID) =================
  zero_f32<<<gs(N_NODES * HID), TB, 0, stream>>>(agg, N_NODES * HID);
  scatter_add<<<8192, TB, 0, stream>>>(h1, src, dst, agg, HID / 4);
  add_cvt_f16<<<gs(N_NODES * HID), TB, 0, stream>>>(h1, agg, zh, N_NODES * HID);
  gemm_wmma<0><<<gemm_blocks, TB, 0, stream>>>(zh, Wh2a, b2a, th, nullptr,
                                               N_NODES, HID, HID);

  // ===== last GEMM fused with global mean-pool numerator =====
  zero_f32<<<gs(NUM_GRAPHS * HID), TB, 0, stream>>>(sums, NUM_GRAPHS * HID);
  zero_f32<<<1, TB, 0, stream>>>(cnts, NUM_GRAPHS);
  count_nodes<<<gs(N_NODES), TB, 0, stream>>>(batc, cnts);
  gemm_wmma<2><<<gemm_blocks, TB, 0, stream>>>(th, Wh2b, b2b, sums, batc,
                                               N_NODES, HID, HID);

  // ===== classifier head =====
  classify<<<(NUM_GRAPHS * NUM_CLASSES + TB - 1) / TB, TB, 0, stream>>>(
      sums, cnts, Wc, bc, out);
}
